// EventSequenceDurationGraphConvModel_8022998909608
// MI455X (gfx1250) — matrix operand order repaired and checked by reading, not compiled, not run on hardware
//
#include <hip/hip_runtime.h>

// ---------------------------------------------------------------------------
// Types for CDNA5 WMMA (wave32): v_wmma_f32_16x16x32_bf16
// ---------------------------------------------------------------------------
typedef __attribute__((ext_vector_type(16))) __bf16 v16bf;
typedef __attribute__((ext_vector_type(8)))  float  v8f;
typedef __attribute__((ext_vector_type(4)))  unsigned int u32x4;

union FragU {
    v16bf v;
    u32x4 q[2];
};

// ---------------------------------------------------------------------------
// bf16 helpers (bf16 stored as unsigned short)
// ---------------------------------------------------------------------------
__device__ inline unsigned short f2bf(float f) {
    unsigned u = __float_as_uint(f);
    unsigned r = u + 0x7FFFu + ((u >> 16) & 1u);   // round-to-nearest-even
    return (unsigned short)(r >> 16);
}

// ---------------------------------------------------------------------------
// Fragment loads matching ISA 7.12.2 VGPR layouts.
// A: 16x32 bf16 (M x K), ldA == K. lane m = lane&15, half = lane>>4.
//    vector elems 0..7  -> K = k0 + 8*half + (0..7)     (16 contiguous bytes)
//    vector elems 8..15 -> K = k0 + 16 + 8*half + (0..7)
// B: 32x16 bf16 (K x N) from pre-transposed weight Wt[Fo x K]:
//    lane n = lane&15, same K order as A (elems 0..7 -> k0+8*half+(0..7), 8..15 -> k0+16+8*half+(0..7))
// ---------------------------------------------------------------------------
__device__ inline v16bf load_frag_a(const unsigned short* __restrict__ A,
                                    long row, int K, int k0, int half) {
    FragU f;
    const unsigned short* p = A + row * (long)K + k0 + 8 * half;
    f.q[0] = *(const u32x4*)(p);
    f.q[1] = *(const u32x4*)(p + 16);
    return f.v;
}

__device__ inline v16bf load_frag_b(const unsigned short* __restrict__ Wt,
                                    int col, int K, int k0, int half) {
    FragU f;
    const unsigned short* p = Wt + (long)col * K + k0 + 8 * half;
    f.q[0] = *(const u32x4*)(p);
    f.q[1] = *(const u32x4*)(p + 16);
    return f.v;
}

// ---------------------------------------------------------------------------
// Dual-input WMMA GEMM, register-resident weights, double-buffered A loads:
//   out[n, colOff + f] = act( A1[n,:K1]@W1 + (K2 ? A2[n,:K2]@W2 : 0) + bias[f] )
// nRows must be a multiple of 16 (true: N=50000, B=64).
// ---------------------------------------------------------------------------
template <int K1, int K2, bool OF, bool OBF, bool RELU>
__global__ void k_gemm_wmma(const unsigned short* __restrict__ A1,
                            const unsigned short* __restrict__ W1t,
                            const unsigned short* __restrict__ A2,
                            const unsigned short* __restrict__ W2t,
                            const float* __restrict__ bias,
                            float* __restrict__ outF,
                            unsigned short* __restrict__ outBF,
                            int nRowTiles, int Fo, int ldOut, int colOff) {
    constexpr int NB1 = K1 / 32;
    constexpr int NB2 = (K2 > 0) ? K2 / 32 : 1;

    const int lane = threadIdx.x;
    const int colTile = blockIdx.y * blockDim.y + threadIdx.y;
    if (colTile >= (Fo >> 4)) return;               // uniform per wave
    const int half = lane >> 4;
    const int mlane = lane & 15;
    const int ncol = colTile * 16 + mlane;

    // Preload all weight (B) fragments into registers.
    v16bf b1[NB1];
#pragma unroll
    for (int i = 0; i < NB1; i++) b1[i] = load_frag_b(W1t, ncol, K1, i * 32, half);
    v16bf b2[NB2];
    if (K2 > 0) {
#pragma unroll
        for (int i = 0; i < NB2; i++) b2[i] = load_frag_b(W2t, ncol, K2, i * 32, half);
    }

    const float bv = bias[ncol];

    for (long rowTile = blockIdx.x; rowTile < nRowTiles; rowTile += gridDim.x) {
        const long row = rowTile * 16 + mlane;
        v8f acc = {};
        // Software pipeline: prefetch fragment i+1 before the WMMA on fragment i,
        // chained across the A1 -> A2 boundary, so loads overlap the matrix pipe.
        v16bf a = load_frag_a(A1, row, K1, 0, half);
#pragma unroll
        for (int i = 0; i < NB1; i++) {
            v16bf an = a;
            if (i + 1 < NB1)      an = load_frag_a(A1, row, K1, (i + 1) * 32, half);
            else if (K2 > 0)      an = load_frag_a(A2, row, K2, 0, half);
            acc = __builtin_amdgcn_wmma_f32_16x16x32_bf16(false, a, false, b1[i],
                                                          (short)0, acc, false, false);
            a = an;
        }
        if (K2 > 0) {
#pragma unroll
            for (int i = 0; i < NB2; i++) {
                v16bf an = a;
                if (i + 1 < NB2)  an = load_frag_a(A2, row, K2, (i + 1) * 32, half);
                acc = __builtin_amdgcn_wmma_f32_16x16x32_bf16(false, a, false, b2[i],
                                                              (short)0, acc, false, false);
                a = an;
            }
        }
#pragma unroll
        for (int r = 0; r < 8; r++) {
            long rr = rowTile * 16 + r + 8 * half;  // C/D layout: VGPR r -> M = r + 8*half
            float v = acc[r] + bv;
            if (RELU) v = fmaxf(v, 0.0f);
            long oidx = rr * (long)ldOut + colOff + ncol;
            if (OF)  outF[oidx]  = v;
            if (OBF) outBF[oidx] = f2bf(v);
        }
    }
}

// ---------------------------------------------------------------------------
// Edge scatter: agg[dst[e], :] += ew[e] * h_bf16[src[e], :]   (f32 atomics)
// One thread per (edge, 4-feature chunk).
// ---------------------------------------------------------------------------
__global__ void k_scatter(const unsigned short* __restrict__ h,
                          const int* __restrict__ src, const int* __restrict__ dst,
                          const float* __restrict__ ew,
                          float* __restrict__ agg, int F, int E) {
    const int chunks = F >> 2;
    long tid = (long)blockIdx.x * blockDim.x + threadIdx.x;
    if (tid >= (long)E * chunks) return;
    int e = (int)(tid / chunks);
    int c = (int)(tid % chunks) * 4;
    int s = src[e], d = dst[e];
    float w = ew[e];
    uint2 a = *(const uint2*)(h + (long)s * F + c);
    float v0 = __uint_as_float((a.x & 0xFFFFu) << 16) * w;
    float v1 = __uint_as_float(a.x & 0xFFFF0000u) * w;
    float v2 = __uint_as_float((a.y & 0xFFFFu) << 16) * w;
    float v3 = __uint_as_float(a.y & 0xFFFF0000u) * w;
    float* ap = agg + (long)d * F + c;
    atomicAdd(ap + 0, v0);
    atomicAdd(ap + 1, v1);
    atomicAdd(ap + 2, v2);
    atomicAdd(ap + 3, v3);
}

// ---------------------------------------------------------------------------
// Small elementwise / conversion kernels
// ---------------------------------------------------------------------------
__global__ void k_mask_bf(const float* __restrict__ x, unsigned short* __restrict__ o, long n) {
    long i = (long)blockIdx.x * blockDim.x + threadIdx.x;
    if (i >= n) return;
    float v = x[i];
    o[i] = f2bf(v == -1.0f ? 0.0f : v);
}

__global__ void k_f2bf(const float* __restrict__ x, unsigned short* __restrict__ o, long n) {
    long i = (long)blockIdx.x * blockDim.x + threadIdx.x;
    if (i >= n) return;
    o[i] = f2bf(x[i]);
}

// Wt[n*K + k] = bf16(W[k*Fo + n])  (transpose + convert)
__global__ void k_transpose_bf(const float* __restrict__ W, unsigned short* __restrict__ Wt,
                               int K, int Fo) {
    int tid = blockIdx.x * blockDim.x + threadIdx.x;
    if (tid >= K * Fo) return;
    int k = tid / Fo, n = tid % Fo;
    Wt[(long)n * K + k] = f2bf(W[tid]);
}

// Wt[n*K + k] = bf16(Wa[k*Fo + n] + Wb[k*Fo + n])
__global__ void k_comb_transpose_bf(const float* __restrict__ Wa, const float* __restrict__ Wb,
                                    unsigned short* __restrict__ Wt, int K, int Fo) {
    int tid = blockIdx.x * blockDim.x + threadIdx.x;
    if (tid >= K * Fo) return;
    int k = tid / Fo, n = tid % Fo;
    Wt[(long)n * K + k] = f2bf(Wa[tid] + Wb[tid]);
}

__global__ void k_add_bias(const float* __restrict__ a, const float* __restrict__ b,
                           float* __restrict__ o, int n) {
    int i = blockIdx.x * blockDim.x + threadIdx.x;
    if (i < n) o[i] = a[i] + b[i];
}

// ---------------------------------------------------------------------------
// Global mean pool: sums[batch[i], :] += xc[i, :]; cnt[batch[i]] += 1
// ---------------------------------------------------------------------------
__global__ void k_pool(const float* __restrict__ xc, const int* __restrict__ batch,
                       float* __restrict__ sums, float* __restrict__ cnt, int N) {
    long tid = (long)blockIdx.x * blockDim.x + threadIdx.x;
    if (tid >= (long)N * 64) return;
    int node = (int)(tid >> 6);
    int c = (int)(tid & 63) * 4;
    int b = batch[node];
    float4 v = *(const float4*)(xc + (long)node * 256 + c);
    float* sp = sums + b * 256 + c;
    atomicAdd(sp + 0, v.x);
    atomicAdd(sp + 1, v.y);
    atomicAdd(sp + 2, v.z);
    atomicAdd(sp + 3, v.w);
    if (c == 0) atomicAdd(&cnt[b], 1.0f);
}

// hcat[b*384 + c] = bf16(sums[b*256+c] / max(cnt[b],1))
__global__ void k_mean_bf(const float* __restrict__ sums, const float* __restrict__ cnt,
                          unsigned short* __restrict__ hcat) {
    int tid = blockIdx.x * blockDim.x + threadIdx.x;
    if (tid >= 64 * 256) return;
    int b = tid >> 8, c = tid & 255;
    float v = sums[tid] / fmaxf(cnt[b], 1.0f);
    hcat[b * 384 + c] = f2bf(v);
}

__global__ void k_logsoftmax(const float* __restrict__ logits, float* __restrict__ out, int B) {
    int b = blockIdx.x * blockDim.x + threadIdx.x;
    if (b >= B) return;
    float m = -1e30f;
    for (int j = 0; j < 16; j++) m = fmaxf(m, logits[b * 16 + j]);
    float s = 0.0f;
    for (int j = 0; j < 16; j++) s += __expf(logits[b * 16 + j] - m);
    float ls = __logf(s);
    for (int j = 0; j < 16; j++) out[b * 16 + j] = logits[b * 16 + j] - m - ls;
}

// ---------------------------------------------------------------------------
// Host orchestration
// ---------------------------------------------------------------------------
extern "C" void kernel_launch(void* const* d_in, const int* in_sizes, int n_in,
                              void* d_out, int out_size, void* d_ws, size_t ws_size,
                              hipStream_t stream) {
    (void)in_sizes; (void)n_in; (void)out_size; (void)ws_size;
    const int N = 50000, E = 800000, B = 64;
    const int NT = N / 16;      // 3125 row tiles (exact)
    const int BT = B / 16;      // 4 row tiles (exact)

    // ---- inputs (setup_inputs order) ----
    const float* x          = (const float*)d_in[0];
    const int*   ei         = (const int*)d_in[1];
    const float* eattr      = (const float*)d_in[2];
    const int*   batch      = (const int*)d_in[3];
    const float* dur_x      = (const float*)d_in[4];
    const int*   dei        = (const int*)d_in[5];
    const float* deattr     = (const float*)d_in[6];
    const float* seq        = (const float*)d_in[7];
    const float* g1_Wr      = (const float*)d_in[8];
    const float* g1_br      = (const float*)d_in[9];
    const float* g1_Wroot   = (const float*)d_in[10];
    const float* g2_Wr      = (const float*)d_in[11];
    const float* g2_br      = (const float*)d_in[12];
    const float* g2_Wroot   = (const float*)d_in[13];
    const float* d1_Wr      = (const float*)d_in[14];
    const float* d1_br      = (const float*)d_in[15];
    const float* d1_Wroot   = (const float*)d_in[16];
    const float* c1_Wr      = (const float*)d_in[17];
    const float* c1_br      = (const float*)d_in[18];
    const float* c1_Wroot   = (const float*)d_in[19];
    const float* skip_W     = (const float*)d_in[20];
    const float* skip_b     = (const float*)d_in[21];
    const float* fc1_W      = (const float*)d_in[22];
    const float* fc1_b      = (const float*)d_in[23];
    const float* fc2_W      = (const float*)d_in[24];
    const float* fc2_b      = (const float*)d_in[25];
    const float* fcc_W      = (const float*)d_in[26];
    const float* fcc_b      = (const float*)d_in[27];
    const float* cls_W      = (const float*)d_in[28];
    const float* cls_b      = (const float*)d_in[29];

    const int* src = ei;        const int* dst = ei + E;
    const int* dsrc = dei;      const int* ddst = dei + E;

    // ---- workspace carve-up ----
    char* wsb = (char*)d_ws;
    size_t off = 0;
    auto alloc = [&](size_t bytes) -> char* {
        char* p = wsb + off;
        off = (off + bytes + 255) & ~(size_t)255;
        return p;
    };
    unsigned short* x_bf   = (unsigned short*)alloc((size_t)N * 128 * 2);
    unsigned short* f1_bf  = (unsigned short*)alloc((size_t)N * 128 * 2);
    unsigned short* xc_bf  = (unsigned short*)alloc((size_t)N * 256 * 2);
    unsigned short* dx_bf  = (unsigned short*)alloc((size_t)N * 64 * 2);
    float*          agg_f  = (float*)alloc((size_t)N * 256 * 4);
    unsigned short* agg_bf = (unsigned short*)alloc((size_t)N * 256 * 2);
    float*          out_f  = (float*)alloc((size_t)N * 256 * 4);
    float*          sums   = (float*)alloc((size_t)B * 256 * 4);
    float*          cntv   = (float*)alloc((size_t)B * 4);
    unsigned short* hcat   = (unsigned short*)alloc((size_t)B * 384 * 2);
    unsigned short* s1_bf  = (unsigned short*)alloc((size_t)B * 256 * 2);
    unsigned short* h2_bf  = (unsigned short*)alloc((size_t)B * 256 * 2);
    unsigned short* seq_bf = (unsigned short*)alloc((size_t)B * 256 * 2);
    float*          logits = (float*)alloc((size_t)B * 16 * 4);
    float*          bcomb  = (float*)alloc(256 * 4);
    unsigned short* g1Wr_t    = (unsigned short*)alloc(128 * 128 * 2);
    unsigned short* g1Wroot_t = (unsigned short*)alloc(128 * 128 * 2);
    unsigned short* g2Wr_t    = (unsigned short*)alloc(128 * 128 * 2);
    unsigned short* g2Wroot_t = (unsigned short*)alloc(128 * 128 * 2);
    unsigned short* d1Wr_t    = (unsigned short*)alloc(64 * 128 * 2);
    unsigned short* d1Wroot_t = (unsigned short*)alloc(64 * 128 * 2);
    unsigned short* c1Wr_t    = (unsigned short*)alloc(256 * 256 * 2);
    unsigned short* wcomb_t   = (unsigned short*)alloc(256 * 256 * 2);
    unsigned short* fc1_t     = (unsigned short*)alloc(256 * 256 * 2);
    unsigned short* fc2_t     = (unsigned short*)alloc(256 * 128 * 2);
    unsigned short* fcc_t     = (unsigned short*)alloc(384 * 256 * 2);
    unsigned short* cls_t     = (unsigned short*)alloc(256 * 16 * 2);

    auto cdiv = [](long a, long b) { return (int)((a + b - 1) / b); };

    // ---- weight transpose+convert ----
    auto tr = [&](const float* W, unsigned short* Wt, int K, int Fo) {
        k_transpose_bf<<<cdiv((long)K * Fo, 256), 256, 0, stream>>>(W, Wt, K, Fo);
    };
    tr(g1_Wr, g1Wr_t, 128, 128);       tr(g1_Wroot, g1Wroot_t, 128, 128);
    tr(g2_Wr, g2Wr_t, 128, 128);       tr(g2_Wroot, g2Wroot_t, 128, 128);
    tr(d1_Wr, d1Wr_t, 64, 128);        tr(d1_Wroot, d1Wroot_t, 64, 128);
    tr(c1_Wr, c1Wr_t, 256, 256);
    k_comb_transpose_bf<<<cdiv(256 * 256, 256), 256, 0, stream>>>(c1_Wroot, skip_W, wcomb_t, 256, 256);
    k_add_bias<<<1, 256, 0, stream>>>(c1_br, skip_b, bcomb, 256);
    tr(fc1_W, fc1_t, 256, 256);
    tr(fc2_W, fc2_t, 256, 128);
    tr(fcc_W, fcc_t, 384, 256);
    tr(cls_W, cls_t, 256, 16);

    // ---- input conversions ----
    k_mask_bf<<<cdiv((long)N * 128, 256), 256, 0, stream>>>(x, x_bf, (long)N * 128);
    k_f2bf<<<cdiv((long)N * 64, 256), 256, 0, stream>>>(dur_x, dx_bf, (long)N * 64);
    k_f2bf<<<cdiv((long)B * 256, 256), 256, 0, stream>>>(seq, seq_bf, (long)B * 256);

    auto scatter = [&](const unsigned short* h, const int* s, const int* d,
                       const float* w, float* agg, int F) {
        long total = (long)E * (F >> 2);
        k_scatter<<<cdiv(total, 256), 256, 0, stream>>>(h, s, d, w, agg, F, E);
    };

    const dim3 blk(32, 4);
    auto grd = [&](int rowTiles, int Fo) {
        return dim3((unsigned)cdiv(rowTiles, 8), (unsigned)cdiv(Fo >> 4, 4));
    };

    // ---- g1: f1 = relu(agg(x)@Wr + x@Wroot + br) ----
    hipMemsetAsync(agg_f, 0, (size_t)N * 128 * 4, stream);
    scatter(x_bf, src, dst, eattr, agg_f, 128);
    k_f2bf<<<cdiv((long)N * 128, 256), 256, 0, stream>>>(agg_f, agg_bf, (long)N * 128);
    k_gemm_wmma<128, 128, false, true, true><<<grd(NT, 128), blk, 0, stream>>>(
        agg_bf, g1Wr_t, x_bf, g1Wroot_t, g1_br, nullptr, f1_bf, NT, 128, 128, 0);

    // ---- g2 -> xc[:, 0:128] ----
    hipMemsetAsync(agg_f, 0, (size_t)N * 128 * 4, stream);
    scatter(f1_bf, src, dst, eattr, agg_f, 128);
    k_f2bf<<<cdiv((long)N * 128, 256), 256, 0, stream>>>(agg_f, agg_bf, (long)N * 128);
    k_gemm_wmma<128, 128, false, true, true><<<grd(NT, 128), blk, 0, stream>>>(
        agg_bf, g2Wr_t, f1_bf, g2Wroot_t, g2_br, nullptr, xc_bf, NT, 128, 256, 0);

    // ---- duration branch -> xc[:, 128:256] ----
    hipMemsetAsync(agg_f, 0, (size_t)N * 64 * 4, stream);
    scatter(dx_bf, dsrc, ddst, deattr, agg_f, 64);
    k_f2bf<<<cdiv((long)N * 64, 256), 256, 0, stream>>>(agg_f, agg_bf, (long)N * 64);
    k_gemm_wmma<64, 64, false, true, true><<<grd(NT, 128), blk, 0, stream>>>(
        agg_bf, d1Wr_t, dx_bf, d1Wroot_t, d1_br, nullptr, xc_bf, NT, 128, 256, 128);

    // ---- c1 + skip fused: out = relu(agg(xc)@c1_Wr + xc@(c1_Wroot+skip_W) + (c1_br+skip_b)) ----
    hipMemsetAsync(agg_f, 0, (size_t)N * 256 * 4, stream);
    scatter(xc_bf, src, dst, eattr, agg_f, 256);
    k_f2bf<<<cdiv((long)N * 256, 256), 256, 0, stream>>>(agg_f, agg_bf, (long)N * 256);
    k_gemm_wmma<256, 256, true, false, true><<<grd(NT, 256), blk, 0, stream>>>(
        agg_bf, c1Wr_t, xc_bf, wcomb_t, bcomb, out_f, nullptr, NT, 256, 256, 0);

    // ---- global mean pool -> hcat[:, 0:256] ----
    hipMemsetAsync(sums, 0, (size_t)B * 256 * 4, stream);
    hipMemsetAsync(cntv, 0, (size_t)B * 4, stream);
    k_pool<<<cdiv((long)N * 64, 256), 256, 0, stream>>>(out_f, batch, sums, cntv, N);
    k_mean_bf<<<cdiv(B * 256, 256), 256, 0, stream>>>(sums, cntv, hcat);

    // ---- sequence MLP -> hcat[:, 256:384] ----
    k_gemm_wmma<256, 0, false, true, true><<<grd(BT, 256), blk, 0, stream>>>(
        seq_bf, fc1_t, nullptr, nullptr, fc1_b, nullptr, s1_bf, BT, 256, 256, 0);
    k_gemm_wmma<256, 0, false, true, true><<<grd(BT, 128), blk, 0, stream>>>(
        s1_bf, fc2_t, nullptr, nullptr, fc2_b, nullptr, hcat, BT, 128, 384, 256);

    // ---- fused head ----
    k_gemm_wmma<384, 0, false, true, true><<<grd(BT, 256), blk, 0, stream>>>(
        hcat, fcc_t, nullptr, nullptr, fcc_b, nullptr, h2_bf, BT, 256, 256, 0);
    k_gemm_wmma<256, 0, true, false, false><<<grd(BT, 16), blk, 0, stream>>>(
        h2_bf, cls_t, nullptr, nullptr, cls_b, logits, nullptr, BT, 16, 16, 0);

    k_logsoftmax<<<cdiv(B, 64), 64, 0, stream>>>(logits, (float*)d_out, B);
}
